// GCN_94489280574
// MI455X (gfx1250) — compile-verified, matrix operand order checked by reading, not run on hardware
//
#include <hip/hip_runtime.h>
#include <hip/hip_bf16.h>
#include <math.h>

typedef __attribute__((ext_vector_type(16))) _Float16 v16h;
typedef __attribute__((ext_vector_type(8)))  float    v8f;

static __device__ __forceinline__ int imin(int a, int b) { return a < b ? a : b; }

// ---------------- degree / symmetric norm ----------------
__global__ void k_deg_init(float* deg, int n) {
    int i = blockIdx.x * blockDim.x + threadIdx.x;
    if (i < n) deg[i] = 1.0f;  // self-loop contributes 1
}

__global__ void k_deg_accum(const long long* __restrict__ ei, float* deg, int E) {
    int e = blockIdx.x * blockDim.x + threadIdx.x;
    if (e < E) {
        int d = (int)ei[(size_t)E + e];  // dst row
        atomicAdd(&deg[d], 1.0f);
    }
}

__global__ void k_deg_finish(float* deg, int n) {
    int i = blockIdx.x * blockDim.x + threadIdx.x;
    if (i < n) deg[i] = rsqrtf(deg[i]);  // deg >= 1 (self-loops)
}

// ---------------- pre-swizzle W[K,64] f32 -> fragment-major f16 ----------------
// Fragment order: frag = kc*NT + t; within a frag, 32 lanes x 16 f16 (32B/lane),
// matching the per-lane B operand of v_wmma_f32_16x16x32_f16:
//   lane L: N = t*16 + (L&15), K = kc*32 + 8*(L>>4) + {0..7, 16..23}
template <int K, int NT>
__global__ void k_swizzle_w(const float* __restrict__ W, _Float16* __restrict__ Wsw) {
    constexpr int NFRAG = (K / 32) * NT;
    int idx = blockIdx.x * blockDim.x + threadIdx.x;  // one thread per (frag, lane)
    if (idx >= NFRAG * 32) return;
    const int lane = idx & 31;
    const int frag = idx >> 5;
    const int t  = frag % NT;
    const int kc = frag / NT;
    const int ml = lane & 15;
    const int kb = (lane >> 4) * 8;
    const int col = t * 16 + ml;
    _Float16* dst = Wsw + (size_t)idx * 16;
#pragma unroll
    for (int j = 0; j < 8; ++j) {
        dst[j]     = (_Float16)W[(size_t)(kc * 32 + kb + j)      * (NT * 16) + col];
        dst[j + 8] = (_Float16)W[(size_t)(kc * 32 + kb + 16 + j) * (NT * 16) + col];
    }
}

// ---------------- WMMA GEMM + fused self-loop init ----------------
// H[n,64] = X[n,K] @ W ;  Agg[n,64] = H * dis^2   (self-loop term of GCN agg)
// One wave computes a 16x64 tile; 4 waves/block; W staged in LDS.
template <int K, int NT>
__global__ void __launch_bounds__(128) k_gemm_wmma(const float* __restrict__ X,
                                                   const _Float16* __restrict__ Wsw,
                                                   const float* __restrict__ dis,
                                                   float* __restrict__ H,
                                                   float* __restrict__ Agg, int n) {
    constexpr int NFRAG = (K / 32) * NT;
    __shared__ v16h sW[NFRAG * 32];

    // cooperative stage of swizzled W into LDS (32B vectors -> b128 pairs)
    const v16h* gW = (const v16h*)Wsw;
    for (int i = threadIdx.x; i < NFRAG * 32; i += 128) sW[i] = gW[i];
    __syncthreads();

    const int lane = threadIdx.x & 31;
    const int wave = threadIdx.x >> 5;
    const int mtiles = (n + 15) >> 4;
    const int tile = blockIdx.x * 4 + wave;
    if (tile >= mtiles) return;  // wave-uniform

    const int row0 = tile * 16;
    const int ml = lane & 15;
    const int hi = lane >> 4;
    const int kb = hi * 8;

    v8f acc[NT] = {};

    const int mrow = imin(row0 + ml, n - 1);
    const float* __restrict__ xrow = X + (size_t)mrow * K;

#pragma unroll
    for (int kc = 0; kc < K / 32; ++kc) {
        // A fragment: 4x float4 loads (2 contiguous 32B runs), convert to f16
        const float4 p0 = *(const float4*)(xrow + kc * 32 + kb);
        const float4 p1 = *(const float4*)(xrow + kc * 32 + kb + 4);
        const float4 p2 = *(const float4*)(xrow + kc * 32 + kb + 16);
        const float4 p3 = *(const float4*)(xrow + kc * 32 + kb + 20);
        v16h a;
        a[0]  = (_Float16)p0.x; a[1]  = (_Float16)p0.y; a[2]  = (_Float16)p0.z; a[3]  = (_Float16)p0.w;
        a[4]  = (_Float16)p1.x; a[5]  = (_Float16)p1.y; a[6]  = (_Float16)p1.z; a[7]  = (_Float16)p1.w;
        a[8]  = (_Float16)p2.x; a[9]  = (_Float16)p2.y; a[10] = (_Float16)p2.z; a[11] = (_Float16)p2.w;
        a[12] = (_Float16)p3.x; a[13] = (_Float16)p3.y; a[14] = (_Float16)p3.z; a[15] = (_Float16)p3.w;
#pragma unroll
        for (int t = 0; t < NT; ++t) {
            const v16h b = sW[(kc * NT + t) * 32 + lane];
            acc[t] = __builtin_amdgcn_wmma_f32_16x16x32_f16(
                false, a, false, b, (short)0, acc[t], false, false);
        }
    }

    // per-row dis^2 for the fused self-loop term (broadcast loads)
    float ds2[8];
#pragma unroll
    for (int v = 0; v < 8; ++v) {
        int m = imin(row0 + v + hi * 8, n - 1);
        float dv = dis[m];
        ds2[v] = dv * dv;
    }

    // D layout: element v -> M = row0 + v + hi*8, N = t*16 + ml
    if (row0 + 16 <= n) {  // full tile: unguarded stores
#pragma unroll
        for (int t = 0; t < NT; ++t) {
#pragma unroll
            for (int v = 0; v < 8; ++v) {
                const size_t o = (size_t)(row0 + v + hi * 8) * 64 + t * 16 + ml;
                const float val = acc[t][v];
                H[o] = val;
                Agg[o] = val * ds2[v];
            }
        }
    } else {
#pragma unroll
        for (int t = 0; t < NT; ++t) {
#pragma unroll
            for (int v = 0; v < 8; ++v) {
                const int m = row0 + v + hi * 8;
                if (m < n) {
                    const size_t o = (size_t)m * 64 + t * 16 + ml;
                    const float val = acc[t][v];
                    H[o] = val;
                    Agg[o] = val * ds2[v];
                }
            }
        }
    }
}

// ---------------- edge scatter (F=64): agg[dst] += h[src]*dis[src]*dis[dst] ----------------
__global__ void k_edge_scatter64(const long long* __restrict__ ei, const float* __restrict__ dis,
                                 const float* __restrict__ h, float* __restrict__ agg, int E) {
    int gid = blockIdx.x * blockDim.x + threadIdx.x;
    int e = gid >> 4;  // 16 threads per edge, 4 floats each
    int c = gid & 15;
    if (e >= E) return;
    int s = (int)ei[e];
    int d = (int)ei[(size_t)E + e];
    float w = dis[s] * dis[d];
    const float4 v = *reinterpret_cast<const float4*>(h + ((size_t)s << 6) + (c << 2));
    float* o = agg + ((size_t)d << 6) + (c << 2);
    atomicAdd(o + 0, v.x * w);
    atomicAdd(o + 1, v.y * w);
    atomicAdd(o + 2, v.z * w);
    atomicAdd(o + 3, v.w * w);
}

__global__ void k_edge_scatter1(const long long* __restrict__ ei, const float* __restrict__ dis,
                                const float* __restrict__ h, float* __restrict__ agg, int E) {
    int e = blockIdx.x * blockDim.x + threadIdx.x;
    if (e >= E) return;
    int s = (int)ei[e];
    int d = (int)ei[(size_t)E + e];
    atomicAdd(&agg[d], h[s] * dis[s] * dis[d]);
}

// ---------------- activations ----------------
__global__ void k_bias_relu64(const float* __restrict__ agg, const float* __restrict__ b,
                              float* __restrict__ out, int n) {
    int idx = blockIdx.x * blockDim.x + threadIdx.x;
    if (idx < n * 64) out[idx] = fmaxf(agg[idx] + b[idx & 63], 0.0f);
}

// layer 3: h3 = h2 @ W3 (64->1), fused with self-loop init agg3 = h3*dis^2
__global__ void k_dot64_selfloop(const float* __restrict__ h, const float* __restrict__ w3,
                                 const float* __restrict__ dis, float* __restrict__ h3,
                                 float* __restrict__ agg3, int n) {
    int i = blockIdx.x * blockDim.x + threadIdx.x;
    if (i >= n) return;
    const float* r = h + ((size_t)i << 6);
    float s = 0.0f;
#pragma unroll
    for (int k = 0; k < 64; ++k) s = fmaf(r[k], w3[k], s);
    h3[i] = s;
    float dv = dis[i];
    agg3[i] = s * dv * dv;
}

__global__ void k_sigmoid(const float* __restrict__ agg, const float* __restrict__ b3,
                          float* __restrict__ out, int n) {
    int i = blockIdx.x * blockDim.x + threadIdx.x;
    if (i < n) out[i] = 1.0f / (1.0f + expf(-(agg[i] + b3[0])));
}

// ---------------- launch ----------------
extern "C" void kernel_launch(void* const* d_in, const int* in_sizes, int n_in,
                              void* d_out, int out_size, void* d_ws, size_t ws_size,
                              hipStream_t stream) {
    const float*     x  = (const float*)d_in[0];
    const float*     W1 = (const float*)d_in[1];
    const float*     b1 = (const float*)d_in[2];
    const float*     W2 = (const float*)d_in[3];
    const float*     b2 = (const float*)d_in[4];
    const float*     W3 = (const float*)d_in[5];
    const float*     b3 = (const float*)d_in[6];
    const long long* ei = (const long long*)d_in[7];  // int64 [2,E]

    const int n = in_sizes[0] / 128;
    const int E = in_sizes[7] / 2;
    float* out = (float*)d_out;

    // workspace layout (256B-aligned slabs)
    char* ws = (char*)d_ws;
    size_t off = 0;
    auto alloc = [&](size_t bytes) { void* p = ws + off; off += (bytes + 255) & ~(size_t)255; return p; };
    float*    dis  = (float*)alloc((size_t)n * 4);
    float*    bufA = (float*)alloc((size_t)n * 64 * 4);
    float*    bufB = (float*)alloc((size_t)n * 64 * 4);
    _Float16* W1s  = (_Float16*)alloc((size_t)128 * 64 * 2);  // swizzled f16
    _Float16* W2s  = (_Float16*)alloc((size_t)64 * 64 * 2);

    const int B = 256;
    auto g = [](long long work, int blk) { return (unsigned)((work + blk - 1) / blk); };

    // weight pre-swizzle (tiny)
    k_swizzle_w<128, 4><<<g(16 * 32, B), B, 0, stream>>>(W1, W1s);
    k_swizzle_w<64, 4><<<g(8 * 32, B), B, 0, stream>>>(W2, W2s);

    // symmetric normalization coefficients
    k_deg_init<<<g(n, B), B, 0, stream>>>(dis, n);
    k_deg_accum<<<g(E, B), B, 0, stream>>>(ei, dis, E);
    k_deg_finish<<<g(n, B), B, 0, stream>>>(dis, n);

    const int mtiles = (n + 15) >> 4;
    const unsigned gemm_grid = (unsigned)((mtiles + 3) / 4);

    // ---- layer 1: 128 -> 64 ----  (gemm writes H=bufA and Agg=bufB)
    k_gemm_wmma<128, 4><<<gemm_grid, 128, 0, stream>>>(x, W1s, dis, bufA, bufB, n);
    k_edge_scatter64<<<g((long long)E * 16, B), B, 0, stream>>>(ei, dis, bufA, bufB, E);
    k_bias_relu64<<<g((long long)n * 64, B), B, 0, stream>>>(bufB, b1, bufA, n);

    // ---- layer 2: 64 -> 64 ----  (Agg written in-place over input; safe per-wave)
    k_gemm_wmma<64, 4><<<gemm_grid, 128, 0, stream>>>(bufA, W2s, dis, bufB, bufA, n);
    k_edge_scatter64<<<g((long long)E * 16, B), B, 0, stream>>>(ei, dis, bufB, bufA, E);
    k_bias_relu64<<<g((long long)n * 64, B), B, 0, stream>>>(bufA, b2, bufB, n);

    // ---- layer 3: 64 -> 1 ----
    k_dot64_selfloop<<<g(n, B), B, 0, stream>>>(bufB, W3, dis, bufA, bufA + n, n);
    k_edge_scatter1<<<g(E, B), B, 0, stream>>>(ei, dis, bufA, bufA + n, E);
    k_sigmoid<<<g(n, B), B, 0, stream>>>(bufA + n, b3, out, n);
}
